// Model_59476707115284
// MI455X (gfx1250) — matrix operand order repaired and checked by reading, not compiled, not run on hardware
//
#include <hip/hip_runtime.h>
#include <hip/hip_bf16.h>

#define DEV __device__ __forceinline__

static constexpr int B_   = 8;
static constexpr int L_   = 1024;
static constexpr int N_   = 512;
static constexpr int H_   = 8;
static constexpr int D_   = 64;      // N/H
static constexpr int DFF_ = 2048;
static constexpr int MAXHOP = 6;
static constexpr int BL_  = B_ * L_;              // 8192 rows
static constexpr size_t SN_ = (size_t)BL_ * N_;   // 4,194,304 elems

typedef __attribute__((ext_vector_type(16))) __bf16          v16bf;
typedef __attribute__((ext_vector_type(16))) unsigned short  v16u;
typedef __attribute__((ext_vector_type(8)))  unsigned short  v8u;
typedef __attribute__((ext_vector_type(8)))  float           v8f;

// ---- f32 -> bf16 (round-to-nearest-even), integer domain ----
DEV unsigned short f2bf(float f) {
    unsigned u = __builtin_bit_cast(unsigned, f);
    u += 0x7FFFu + ((u >> 16) & 1u);
    return (unsigned short)(u >> 16);
}

DEV v8f wmma_bf16(v16bf a, v16bf b, v8f c) {
    return __builtin_amdgcn_wmma_f32_16x16x32_bf16(
        false, a, false, b, (short)0, c, false, false);
}

// A fragment (16 rows x 32 k) from bf16 memory. Row = lane&15.
// a[i] = base[(lane&15)*ld + (lane>>4)*16 + i]  -> 2 x b128 loads
DEV v16bf load_a_frag(const unsigned short* __restrict__ base, int ld) {
    const int l = threadIdx.x & 31;
    const unsigned short* p = base + (size_t)(l & 15) * ld + ((l >> 4) * 16);
    v8u lo = *(const v8u*)(p);
    v8u hi = *(const v8u*)(p + 8);
    v16u u = __builtin_shufflevector(lo, hi, 0,1,2,3,4,5,6,7,8,9,10,11,12,13,14,15);
    return __builtin_bit_cast(v16bf, u);
}

// B fragment (32 k x 16 cols) from bf16 memory holding [col][k] row-major.
// b[i] = base[(lane&15)*ld + (lane>>4)*16 + i]  -> 2 x b128 loads
DEV v16bf load_b_frag(const unsigned short* __restrict__ base, int ld) {
    const int l = threadIdx.x & 31;
    const unsigned short* p = base + (size_t)(l & 15) * ld + ((l >> 4) * 16);
    v8u lo = *(const v8u*)(p);
    v8u hi = *(const v8u*)(p + 8);
    v16u u = __builtin_shufflevector(lo, hi, 0,1,2,3,4,5,6,7,8,9,10,11,12,13,14,15);
    return __builtin_bit_cast(v16bf, u);
}

// ---------------------------------------------------------------------------
// f32 -> bf16 bulk convert (weights, once per launch)
// ---------------------------------------------------------------------------
__global__ void cvt_bf16_kernel(const float* __restrict__ src,
                                unsigned short* __restrict__ dst, int n) {
    int i = blockIdx.x * blockDim.x + threadIdx.x;
    if (i < n) dst[i] = f2bf(src[i]);
}

// ---------------------------------------------------------------------------
// K1: s = src + pos(l) + pos(t); also emit bf16 copy
// ---------------------------------------------------------------------------
__global__ void add_pos_kernel(const float* __restrict__ src, float* __restrict__ dst,
                               unsigned short* __restrict__ dstb,
                               const int* __restrict__ outer_step, int step) {
    const int row  = blockIdx.x;          // b*L + l
    const int lpos = row & (L_ - 1);
    const int t    = MAXHOP * outer_step[0] + step;
    const float c  = -9.210340371976184f / (float)N_;   // -ln(10000)/N
    for (int n = threadIdx.x; n < N_; n += blockDim.x) {
        int   i2  = n >> 1;
        float div = __expf((float)(2 * i2) * c);
        float a1  = (float)lpos * div;
        float a2  = (float)t * div;
        float pe1 = (n & 1) ? __cosf(a1) : __sinf(a1);
        float pe2 = (n & 1) ? __cosf(a2) : __sinf(a2);
        float v   = src[(size_t)row * N_ + n] + pe1 + pe2;
        dst[(size_t)row * N_ + n]  = v;
        dstb[(size_t)row * N_ + n] = f2bf(v);
    }
}

// ---------------------------------------------------------------------------
// K2: halting gate + ACT bookkeeping. One wave per (b,l) row.
// ---------------------------------------------------------------------------
__global__ void halting_kernel(const float* __restrict__ s,
                               const float* __restrict__ p_w, const float* __restrict__ p_b,
                               float* __restrict__ hp, float* __restrict__ rem,
                               float* __restrict__ uw) {
    const int gw = blockIdx.x * (blockDim.x >> 5) + (threadIdx.x >> 5);
    const int l  = threadIdx.x & 31;
    const float* row = s + (size_t)gw * N_;
    float acc = 0.f;
    for (int n = l; n < N_; n += 32) acc += row[n] * p_w[n];
#pragma unroll
    for (int off = 16; off; off >>= 1) acc += __shfl_xor(acc, off, 32);
    float p = 1.f / (1.f + __expf(-(acc + p_b[0])));
    if (l == 0) {
        float h = hp[gw], r = rem[gw];
        float still0 = (h < 1.0f) ? 1.f : 0.f;
        float cond   = h + p * still0;
        float nh     = ((cond > 0.9f) ? 1.f : 0.f) * still0;
        float st     = ((cond <= 0.9f) ? 1.f : 0.f) * still0;
        h += p * st;
        r += nh * (1.f - h);
        h += nh * r;
        hp[gw] = h; rem[gw] = r;
        uw[gw] = p * st + nh * r;
    }
}

// ---------------------------------------------------------------------------
// K3: C[M,N] = act(A[M,K] @ W[N,K]^T + bias) * oscale
//   A, W are bf16. Wave = 32x64 tile (2x4 accs), block = 128x128 (8 waves).
//   OUTMODE: 0 = f32 row-major, 1 = bf16 row-major, 2 = bf16 (B,H,D,L) transposed
// ---------------------------------------------------------------------------
template <int OUTMODE, int RELU>
__global__ void gemm_kernel(const unsigned short* __restrict__ A,
                            const unsigned short* __restrict__ W,
                            const float* __restrict__ bias,
                            float* __restrict__ Cf, unsigned short* __restrict__ Cb,
                            int M, int N, int K, float oscale) {
    const int warp = threadIdx.x >> 5;
    const int l    = threadIdx.x & 31;
    const int m0   = blockIdx.x * 128 + (warp & 3) * 32;
    const int n0   = blockIdx.y * 128 + (warp >> 2) * 64;
    v8f acc[2][4] = {};
    for (int kk = 0; kk < K; kk += 32) {
        v16bf a0 = load_a_frag(A + (size_t)m0 * K + kk, K);
        v16bf a1 = load_a_frag(A + (size_t)(m0 + 16) * K + kk, K);
        v16bf b0 = load_b_frag(W + (size_t)(n0     ) * K + kk, K);
        v16bf b1 = load_b_frag(W + (size_t)(n0 + 16) * K + kk, K);
        v16bf b2 = load_b_frag(W + (size_t)(n0 + 32) * K + kk, K);
        v16bf b3 = load_b_frag(W + (size_t)(n0 + 48) * K + kk, K);
        if (kk + 32 < K) {  // gfx1250 global_prefetch_b8, one k-step ahead
            __builtin_prefetch(A + (size_t)(m0 + (l & 15)) * K + kk + 32, 0, 0);
            __builtin_prefetch(W + (size_t)(n0 + (l & 31)) * K + kk + 32, 0, 0);
        }
        acc[0][0] = wmma_bf16(a0, b0, acc[0][0]);
        acc[0][1] = wmma_bf16(a0, b1, acc[0][1]);
        acc[0][2] = wmma_bf16(a0, b2, acc[0][2]);
        acc[0][3] = wmma_bf16(a0, b3, acc[0][3]);
        acc[1][0] = wmma_bf16(a1, b0, acc[1][0]);
        acc[1][1] = wmma_bf16(a1, b1, acc[1][1]);
        acc[1][2] = wmma_bf16(a1, b2, acc[1][2]);
        acc[1][3] = wmma_bf16(a1, b3, acc[1][3]);
    }
#pragma unroll
    for (int i = 0; i < 2; ++i)
#pragma unroll
        for (int j = 0; j < 4; ++j) {
            int col = n0 + j * 16 + (l & 15);
            int rb  = m0 + i * 16 + ((l >> 4) * 8);
            float bv = bias[col];
#pragma unroll
            for (int r = 0; r < 8; ++r) {
                float v = acc[i][j][r] + bv;
                if (RELU) v = fmaxf(v, 0.f);
                v *= oscale;
                int row = rb + r;
                if (OUTMODE == 0) {
                    Cf[(size_t)row * N + col] = v;
                } else if (OUTMODE == 1) {
                    Cb[(size_t)row * N + col] = f2bf(v);
                } else {  // V^T per head: (B,H,D,L)
                    int b  = row >> 10, lp = row & (L_ - 1);
                    int h  = col >> 6,  dd = col & (D_ - 1);
                    Cb[(((size_t)b * H_ + h) * D_ + dd) * L_ + lp] = f2bf(v);
                }
            }
        }
}

// ---------------------------------------------------------------------------
// K4: flash attention (bf16 operands). One wave per (b, h, 16-query tile).
// Q pre-scaled by 1/sqrt(d) in its projection. V stored (B,H,D,L).
// Scores computed transposed: S^T = K @ Q^T, softmax over the M (key) dim.
// ---------------------------------------------------------------------------
__global__ void attention_kernel(const unsigned short* __restrict__ Q,
                                 const unsigned short* __restrict__ Kb,
                                 const unsigned short* __restrict__ Vt,
                                 unsigned short* __restrict__ O) {
    const int gw = blockIdx.x * (blockDim.x >> 5) + (threadIdx.x >> 5);
    const int b  = gw >> 9;
    const int h  = (gw >> 6) & 7;
    const int qt = gw & 63;
    const int l  = threadIdx.x & 31;
    const int qbase = qt * 16;

    const unsigned short* Qp = Q + ((size_t)(b * L_ + qbase)) * N_ + h * D_;
    v16bf qf[2];
    qf[0] = load_b_frag(Qp, N_);
    qf[1] = load_b_frag(Qp + 32, N_);

    const unsigned short* Vh = Vt + ((size_t)(b * H_ + h)) * D_ * L_;

    float m = -3.0e38f, ssum = 0.f;
    v8f o[4] = {};

    for (int kb = 0; kb < L_; kb += 32) {
        const unsigned short* Kp = Kb + ((size_t)(b * L_ + kb)) * N_ + h * D_;
        v8f s0 = {}, s1 = {};
#pragma unroll
        for (int c = 0; c < 2; ++c) {
            v16bf ka  = load_a_frag(Kp + c * 32, N_);
            v16bf ka2 = load_a_frag(Kp + (size_t)16 * N_ + c * 32, N_);
            s0 = wmma_bf16(ka,  qf[c], s0);
            s1 = wmma_bf16(ka2, qf[c], s1);
        }
        float lm = -3.0e38f;
#pragma unroll
        for (int j = 0; j < 8; ++j) lm = fmaxf(lm, fmaxf(s0[j], s1[j]));
        lm = fmaxf(lm, __shfl_xor(lm, 16, 32));
        float mn = fmaxf(m, lm);
        float alpha = __expf(m - mn);
        m = mn;
        float p0[8], p1[8], ps = 0.f;
#pragma unroll
        for (int j = 0; j < 8; ++j) {
            p0[j] = __expf(s0[j] - mn);
            p1[j] = __expf(s1[j] - mn);
            ps += p0[j] + p1[j];
        }
        ps += __shfl_xor(ps, 16, 32);
        ssum = ssum * alpha + ps;
#pragma unroll
        for (int t = 0; t < 4; ++t)
#pragma unroll
            for (int r = 0; r < 8; ++r) o[t][r] *= alpha;

        // reshape P^T (32 keys x 16 q) C-layout -> B-fragment layout via one
        // cross-half shuffle (no LDS)
        v16u pu;
        const bool lo = (l < 16);
#pragma unroll
        for (int j = 0; j < 8; ++j) {
            float t0 = __shfl_xor(p0[j], 16, 32);
            float t1 = __shfl_xor(p1[j], 16, 32);
            pu[j]     = f2bf(lo ? p0[j] : t1);
            pu[8 + j] = f2bf(lo ? t0 : p1[j]);
        }
        v16bf pf = __builtin_bit_cast(v16bf, pu);

        // O^T += V^T @ P^T ; V^T tiles are contiguous rows of (B,H,D,L)
#pragma unroll
        for (int t = 0; t < 4; ++t) {
            v16bf vf = load_a_frag(Vh + (size_t)(t * 16) * L_ + kb, L_);
            o[t] = wmma_bf16(vf, pf, o[t]);
        }
    }
    const float inv = 1.f / ssum;
    const int q = qbase + (l & 15);
#pragma unroll
    for (int t = 0; t < 4; ++t)
#pragma unroll
        for (int r = 0; r < 8; ++r) {
            int d = t * 16 + (l >> 4) * 8 + r;
            O[((size_t)(b * L_ + q)) * N_ + h * D_ + d] = f2bf(o[t][r] * inv);
        }
}

// ---------------------------------------------------------------------------
// K5: out = LN(xa + xb) * g + b (f32 + bf16 copies). One wave per row.
// ---------------------------------------------------------------------------
__global__ void add_ln_kernel(const float* __restrict__ xa, const float* __restrict__ xb,
                              const float* __restrict__ g, const float* __restrict__ be,
                              float* __restrict__ out, unsigned short* __restrict__ outb) {
    const int gw = blockIdx.x * (blockDim.x >> 5) + (threadIdx.x >> 5);
    const int l  = threadIdx.x & 31;
    const size_t off = (size_t)gw * N_;
    float vals[16], s1 = 0.f;
#pragma unroll
    for (int i = 0; i < 16; ++i) {
        int idx = l + 32 * i;
        float v = xa[off + idx] + xb[off + idx];
        vals[i] = v; s1 += v;
    }
#pragma unroll
    for (int o2 = 16; o2; o2 >>= 1) s1 += __shfl_xor(s1, o2, 32);
    float mean = s1 * (1.f / N_);
    float s2 = 0.f;
#pragma unroll
    for (int i = 0; i < 16; ++i) { float d = vals[i] - mean; s2 += d * d; }
#pragma unroll
    for (int o2 = 16; o2; o2 >>= 1) s2 += __shfl_xor(s2, o2, 32);
    float rs = rsqrtf(s2 * (1.f / N_) + 1e-5f);
#pragma unroll
    for (int i = 0; i < 16; ++i) {
        int idx = l + 32 * i;
        float y = (vals[i] - mean) * rs * g[idx] + be[idx];
        out[off + idx]  = y;
        outb[off + idx] = f2bf(y);
    }
}

// K6: s = LN(xa + xb); prev = s*uw + prev*(1-uw)
__global__ void add_ln_update_kernel(const float* __restrict__ xa, const float* __restrict__ xb,
                                     const float* __restrict__ g, const float* __restrict__ be,
                                     float* __restrict__ s_out, float* __restrict__ prev,
                                     const float* __restrict__ uw) {
    const int gw = blockIdx.x * (blockDim.x >> 5) + (threadIdx.x >> 5);
    const int l  = threadIdx.x & 31;
    const size_t off = (size_t)gw * N_;
    float vals[16], s1 = 0.f;
#pragma unroll
    for (int i = 0; i < 16; ++i) {
        int idx = l + 32 * i;
        float v = xa[off + idx] + xb[off + idx];
        vals[i] = v; s1 += v;
    }
#pragma unroll
    for (int o2 = 16; o2; o2 >>= 1) s1 += __shfl_xor(s1, o2, 32);
    float mean = s1 * (1.f / N_);
    float s2 = 0.f;
#pragma unroll
    for (int i = 0; i < 16; ++i) { float d = vals[i] - mean; s2 += d * d; }
#pragma unroll
    for (int o2 = 16; o2; o2 >>= 1) s2 += __shfl_xor(s2, o2, 32);
    float rs = rsqrtf(s2 * (1.f / N_) + 1e-5f);
    float u  = uw[gw];
#pragma unroll
    for (int i = 0; i < 16; ++i) {
        int idx = l + 32 * i;
        float y = (vals[i] - mean) * rs * g[idx] + be[idx];
        s_out[off + idx] = y;
        prev[off + idx] = y * u + prev[off + idx] * (1.f - u);
    }
}

// ---------------------------------------------------------------------------
extern "C" void kernel_launch(void* const* d_in, const int* in_sizes, int n_in,
                              void* d_out, int out_size, void* d_ws, size_t ws_size,
                              hipStream_t stream) {
    const float* state = (const float*)d_in[0];
    const int*   ostep = (const int*)  d_in[1];
    const float* p_w = (const float*)d_in[2];  const float* p_b = (const float*)d_in[3];
    const float* wq  = (const float*)d_in[4];  const float* bq  = (const float*)d_in[5];
    const float* wk  = (const float*)d_in[6];  const float* bk  = (const float*)d_in[7];
    const float* wv  = (const float*)d_in[8];  const float* bv  = (const float*)d_in[9];
    const float* wo  = (const float*)d_in[10]; const float* bo  = (const float*)d_in[11];
    const float* w1  = (const float*)d_in[12]; const float* b1  = (const float*)d_in[13];
    const float* w2  = (const float*)d_in[14]; const float* b2  = (const float*)d_in[15];
    const float* ln1g = (const float*)d_in[16]; const float* ln1b = (const float*)d_in[17];
    const float* ln2g = (const float*)d_in[18]; const float* ln2b = (const float*)d_in[19];

    char* cur = (char*)d_ws;
    auto alloc_f = [&](size_t n) { float* p = (float*)cur; cur += n * 4; return p; };
    auto alloc_h = [&](size_t n) { unsigned short* p = (unsigned short*)cur; cur += n * 2; return p; };

    float* s    = alloc_f(SN_);
    float* prev = alloc_f(SN_);
    float* x1   = alloc_f(SN_);
    float* proj = alloc_f(SN_);
    unsigned short* s_bf   = alloc_h(SN_);
    unsigned short* x1_bf  = alloc_h(SN_);
    unsigned short* q_bf   = alloc_h(SN_);
    unsigned short* k_bf   = alloc_h(SN_);
    unsigned short* vt_bf  = alloc_h(SN_);
    unsigned short* at_bf  = alloc_h(SN_);
    unsigned short* hdn_bf = alloc_h((size_t)BL_ * DFF_);
    unsigned short* wq_bf  = alloc_h((size_t)N_ * N_);
    unsigned short* wk_bf  = alloc_h((size_t)N_ * N_);
    unsigned short* wv_bf  = alloc_h((size_t)N_ * N_);
    unsigned short* wo_bf  = alloc_h((size_t)N_ * N_);
    unsigned short* w1_bf  = alloc_h((size_t)N_ * DFF_);
    unsigned short* w2_bf  = alloc_h((size_t)N_ * DFF_);
    float* hp  = alloc_f(BL_);
    float* rem = alloc_f(BL_);
    float* uw  = alloc_f(BL_);

    hipMemsetAsync(prev, 0, SN_ * sizeof(float), stream);
    hipMemsetAsync(hp,   0, BL_ * sizeof(float), stream);
    hipMemsetAsync(rem,  0, BL_ * sizeof(float), stream);

    const dim3 blk(256);
    const int NN = N_ * N_, ND = N_ * DFF_;
    cvt_bf16_kernel<<<(NN + 255) / 256, blk, 0, stream>>>(wq, wq_bf, NN);
    cvt_bf16_kernel<<<(NN + 255) / 256, blk, 0, stream>>>(wk, wk_bf, NN);
    cvt_bf16_kernel<<<(NN + 255) / 256, blk, 0, stream>>>(wv, wv_bf, NN);
    cvt_bf16_kernel<<<(NN + 255) / 256, blk, 0, stream>>>(wo, wo_bf, NN);
    cvt_bf16_kernel<<<(ND + 255) / 256, blk, 0, stream>>>(w1, w1_bf, ND);
    cvt_bf16_kernel<<<(ND + 255) / 256, blk, 0, stream>>>(w2, w2_bf, ND);

    const dim3 g_gemm512(BL_ / 128, N_ / 128);    // 64 x 4
    const dim3 g_gemmDFF(BL_ / 128, DFF_ / 128);  // 64 x 16
    const dim3 g_rows(BL_ / 8);
    const dim3 g_attn((B_ * H_ * (L_ / 16)) / 8); // 512 blocks

    for (int step = 0; step < MAXHOP; ++step) {
        add_pos_kernel<<<BL_, blk, 0, stream>>>(step == 0 ? state : s, s, s_bf, ostep, step);
        halting_kernel<<<g_rows, blk, 0, stream>>>(s, p_w, p_b, hp, rem, uw);
        // Q scaled by 1/sqrt(d)=0.125; K bf16; V stored (B,H,D,L)
        gemm_kernel<1, 0><<<g_gemm512, blk, 0, stream>>>(s_bf, wq_bf, bq, nullptr, q_bf,
                                                         BL_, N_, N_, 0.125f);
        gemm_kernel<1, 0><<<g_gemm512, blk, 0, stream>>>(s_bf, wk_bf, bk, nullptr, k_bf,
                                                         BL_, N_, N_, 1.0f);
        gemm_kernel<2, 0><<<g_gemm512, blk, 0, stream>>>(s_bf, wv_bf, bv, nullptr, vt_bf,
                                                         BL_, N_, N_, 1.0f);
        attention_kernel<<<g_attn, blk, 0, stream>>>(q_bf, k_bf, vt_bf, at_bf);
        gemm_kernel<0, 0><<<g_gemm512, blk, 0, stream>>>(at_bf, wo_bf, bo, proj, nullptr,
                                                         BL_, N_, N_, 1.0f);
        add_ln_kernel<<<g_rows, blk, 0, stream>>>(s, proj, ln1g, ln1b, x1, x1_bf);
        gemm_kernel<1, 1><<<g_gemmDFF, blk, 0, stream>>>(x1_bf, w1_bf, b1, nullptr, hdn_bf,
                                                         BL_, DFF_, N_, 1.0f);
        gemm_kernel<0, 0><<<g_gemm512, blk, 0, stream>>>(hdn_bf, w2_bf, b2, proj, nullptr,
                                                         BL_, N_, DFF_, 1.0f);
        add_ln_update_kernel<<<g_rows, blk, 0, stream>>>(x1, proj, ln2g, ln2b, s, prev, uw);
    }
    hipMemcpyAsync(d_out, prev, SN_ * sizeof(float), hipMemcpyDeviceToDevice, stream);
}
